// WindowAttention_59871844106599
// MI455X (gfx1250) — matrix operand order repaired and checked by reading, not compiled, not run on hardware
//
#include <hip/hip_runtime.h>
#include <hip/hip_bf16.h>

// ---------------------------------------------------------------------------
// Fused Swin window attention for gfx1250 (CDNA5, wave32, WMMA 16x16x32 bf16)
//   B_=4096 windows, N=64 tokens, C=192, H=6 heads, hd=32, nW=4096
// One 256-thread block (8 wave32) per window; everything stays in LDS.
// ---------------------------------------------------------------------------

typedef __bf16 bf16;
typedef bf16  bf16x16 __attribute__((ext_vector_type(16)));
typedef bf16  bf16x4  __attribute__((ext_vector_type(4)));
typedef float f32x8   __attribute__((ext_vector_type(8)));
typedef float f32x4   __attribute__((ext_vector_type(4)));
typedef unsigned short u16x8 __attribute__((ext_vector_type(8)));

union FragU { bf16x16 f; u16x8 u[2]; };

static constexpr int Nt   = 64;    // tokens per window
static constexpr int C    = 192;   // channels
static constexpr int H    = 6;     // heads
static constexpr int HD   = 32;    // head dim
static constexpr int C3   = 576;   // 3*C
static constexpr float SCALE = 0.17677669529663687f; // hd^-0.5

// padded LDS strides (pads keep b128 alignment and spread banks)
static constexpr int XS = C + 8;    // x   [64][200] bf16
static constexpr int QS = HD + 8;   // q/k [6][64][40] bf16
static constexpr int VS = Nt + 8;   // vT  [6][32][72] bf16
static constexpr int SS = Nt + 4;   // sc  [6][64][68] f32 (rows 16B-aligned)
static constexpr int AS = Nt + 8;   // attn[6][64][72] bf16
static constexpr int CS = C + 8;    // ctx [64][200] bf16

// region A: max(x 25600, sc 104448, ctx 25600) = 104448 B
// region B+C: q 30720 + k 30720 (attn 55296 overlays both)
// region D: vT 27648
static constexpr int OFF_Q  = 104448;
static constexpr int OFF_K  = OFF_Q + 30720;
static constexpr int OFF_VT = OFF_K + 30720;
static constexpr int LDS_BYTES = OFF_VT + 27648;   // 193536 B < 320 KB/WGP

// ---- WMMA fragment loaders (contiguous k order, same in A and B) --------------------
// A (16x32, 16-bit): lane<16 -> row=lane, elems K=0..15;
//                    lane>=16 -> row=lane-16, elems K=16..31
__device__ __forceinline__ bf16x16 load_frag_a(const bf16* base, int stride, int lane) {
    const int r = lane & 15;
    const int h = (lane >> 4) << 4;          // 0 or 16
    FragU fr;
    fr.u[0] = *(const u16x8*)(base + r * stride + h);
    fr.u[1] = *(const u16x8*)(base + r * stride + h + 8);
    return fr.f;
}
// B (32x16, 16-bit) stored K-major ([col][K]): lane<16 -> col=lane, K=0..15;
//                                              lane>=16 -> col=lane-16, K=16..31
__device__ __forceinline__ bf16x16 load_frag_b(const bf16* base, int stride, int lane) {
    const int c  = lane & 15;
    const int ks = (lane >> 4) << 4;         // 0 or 16
    FragU fr;
    fr.u[0] = *(const u16x8*)(base + c * stride + ks);
    fr.u[1] = *(const u16x8*)(base + c * stride + ks + 8);
    return fr.f;
}

__device__ __forceinline__ f32x8 wmma_bf16(bf16x16 a, bf16x16 b, f32x8 c) {
    return __builtin_amdgcn_wmma_f32_16x16x32_bf16(
        /*neg_a=*/false, a, /*neg_b=*/false, b,
        /*c_mod=*/(short)0, c, /*reuse_a=*/false, /*reuse_b=*/false);
}

// ---------------------------------------------------------------------------
__global__ void wa_convert_weights(const float* __restrict__ wq,
                                   const float* __restrict__ wp,
                                   bf16* __restrict__ wq_bf,
                                   bf16* __restrict__ wp_bf) {
    const int i = blockIdx.x * blockDim.x + threadIdx.x;
    if (i < C3 * C)  wq_bf[i] = (bf16)wq[i];
    if (i < C * C)   wp_bf[i] = (bf16)wp[i];
}

// ---------------------------------------------------------------------------
__global__ __launch_bounds__(256)
void wa_window_attention(const float* __restrict__ x,
                         const float* __restrict__ mask,
                         const bf16*  __restrict__ wqkv,  // [576][192] K-major
                         const float* __restrict__ bqkv,  // [576]
                         const bf16*  __restrict__ wproj, // [192][192] K-major
                         const float* __restrict__ bproj, // [192]
                         float* __restrict__ out) {
    extern __shared__ char smem[];
    bf16*  xl   = (bf16*) smem;               // region A: x [64][XS]
    float* sc   = (float*)smem;               // region A: scores [6][64][SS]
    bf16*  ctx  = (bf16*) smem;               // region A: ctx [64][CS]
    bf16*  ql   = (bf16*)(smem + OFF_Q);      // region B: q [6][64][QS]
    bf16*  kl   = (bf16*)(smem + OFF_K);      // region C: k [6][64][QS]
    bf16*  attn = ql;                         // regions B+C: attn [6][64][AS]
    bf16*  vt   = (bf16*)(smem + OFF_VT);     // region D: vT [6][32][VS]

    const int tid  = threadIdx.x;
    const int lane = tid & 31;
    const int wv   = tid >> 5;                 // wave id 0..7
    const int w    = blockIdx.x;               // window id

    const float* mg = mask + (size_t)w * (Nt * Nt);
    // CDNA5 global_prefetch_b8: pull this window's 16KB mask toward L2/L0 early
    __builtin_prefetch(mg + tid * 16, 0, 0);

    // ---- stage x -> LDS (bf16), vectorized: row = tid>>2, 4 col chunks ----
    {
        const float* xg = x + (size_t)w * (Nt * C);
        const int row = tid >> 2;
        const int cb  = (tid & 3) * 48;
        #pragma unroll
        for (int it = 0; it < 12; ++it) {
            const int col = cb + it * 4;
            const f32x4 v = *(const f32x4*)(xg + row * C + col);
            bf16x4 b; b[0] = (bf16)v[0]; b[1] = (bf16)v[1];
                      b[2] = (bf16)v[2]; b[3] = (bf16)v[3];
            *(bf16x4*)(xl + row * XS + col) = b;
        }
    }
    __syncthreads();

    const int nn  = lane & 15;
    const int hi8 = (lane >> 4) << 3;          // row offset 0 or 8 in C/D frags

    // ---- QKV = x @ Wqkv^T + b : 4 x 36 tiles of 16x16, K=192 -------------
    for (int t = wv; t < 4 * 36; t += 8) {
        const int mb = t / 36, ob = t % 36;
        f32x8 acc0 = {}, acc1 = {};
        #pragma unroll
        for (int k0 = 0; k0 < C; k0 += 64) {   // two independent WMMA chains
            acc0 = wmma_bf16(load_frag_a(xl + mb * 16 * XS + k0, XS, lane),
                             load_frag_b(wqkv + (size_t)(ob * 16) * C + k0, C, lane), acc0);
            acc1 = wmma_bf16(load_frag_a(xl + mb * 16 * XS + k0 + 32, XS, lane),
                             load_frag_b(wqkv + (size_t)(ob * 16) * C + k0 + 32, C, lane), acc1);
        }
        const f32x8 acc = acc0 + acc1;
        // wave-uniform routing: force scalar so stores use s_cbranch, not exec masks
        const int which = __builtin_amdgcn_readfirstlane(ob / 12);       // 0=q 1=k 2=v
        const int obl   = __builtin_amdgcn_readfirstlane(ob - which * 12);
        const int hh    = obl >> 1;            // head (uniform per tile)
        const int d     = ((obl & 1) << 4) + nn;  // head-dim
        const float bias = bqkv[ob * 16 + nn];
        #pragma unroll
        for (int j = 0; j < 8; ++j) {
            const int tok = mb * 16 + j + hi8;
            const float v = acc[j] + bias;
            if      (which == 0) ql[hh * (Nt * QS) + tok * QS + d] = (bf16)(v * SCALE);
            else if (which == 1) kl[hh * (Nt * QS) + tok * QS + d] = (bf16)v;
            else                 vt[hh * (HD * VS) + d * VS + tok] = (bf16)v;
        }
    }
    __syncthreads();

    // ---- scores = (q*scale) @ k^T + mask : 6 heads x 4x4 tiles, K=32 -----
    for (int t = wv; t < H * 16; t += 8) {
        const int hh = t >> 4, mb = (t >> 2) & 3, nb = t & 3;
        bf16x16 a = load_frag_a(ql + hh * (Nt * QS) + mb * 16 * QS, QS, lane);
        bf16x16 b = load_frag_b(kl + hh * (Nt * QS) + nb * 16 * QS, QS, lane);
        f32x8 acc = {};
        acc = wmma_bf16(a, b, acc);
        #pragma unroll
        for (int j = 0; j < 8; ++j) {
            const int row = mb * 16 + j + hi8;
            const int col = nb * 16 + nn;
            sc[hh * (Nt * SS) + row * SS + col] = acc[j] + mg[row * Nt + col];
        }
    }
    __syncthreads();

    // ---- row softmax -> bf16 probabilities (overlays dead q/k) -----------
    // single pass: row into registers (16 x ds_load_b128), one exp per elem
    for (int r = tid; r < H * Nt; r += 256) {
        const float* srow = sc + r * SS;       // 16B-aligned rows (SS=68)
        f32x4 buf[16];
        #pragma unroll
        for (int i = 0; i < 16; ++i) buf[i] = *(const f32x4*)(srow + 4 * i);
        float mx = -3.0e38f;
        #pragma unroll
        for (int i = 0; i < 16; ++i) {
            mx = fmaxf(mx, fmaxf(fmaxf(buf[i][0], buf[i][1]),
                                 fmaxf(buf[i][2], buf[i][3])));
        }
        float sum = 0.f;
        #pragma unroll
        for (int i = 0; i < 16; ++i) {
            buf[i][0] = __expf(buf[i][0] - mx); buf[i][1] = __expf(buf[i][1] - mx);
            buf[i][2] = __expf(buf[i][2] - mx); buf[i][3] = __expf(buf[i][3] - mx);
            sum += buf[i][0] + buf[i][1] + buf[i][2] + buf[i][3];
        }
        const float inv = 1.f / sum;
        bf16* arow = attn + r * AS;
        #pragma unroll
        for (int i = 0; i < 16; ++i) {
            bf16x4 b; b[0] = (bf16)(buf[i][0] * inv); b[1] = (bf16)(buf[i][1] * inv);
                      b[2] = (bf16)(buf[i][2] * inv); b[3] = (bf16)(buf[i][3] * inv);
            *(bf16x4*)(arow + 4 * i) = b;
        }
    }
    __syncthreads();

    // ---- ctx = attn @ v : 6 heads x 4x2 tiles, K=64 (overlays dead sc) ---
    for (int t = wv; t < H * 8; t += 8) {
        const int hh = t >> 3, mb = (t >> 1) & 3, db = t & 1;
        f32x8 acc0 = {}, acc1 = {};
        acc0 = wmma_bf16(load_frag_a(attn + hh * (Nt * AS) + mb * 16 * AS, AS, lane),
                         load_frag_b(vt + hh * (HD * VS) + db * 16 * VS, VS, lane), acc0);
        acc1 = wmma_bf16(load_frag_a(attn + hh * (Nt * AS) + mb * 16 * AS + 32, AS, lane),
                         load_frag_b(vt + hh * (HD * VS) + db * 16 * VS + 32, VS, lane), acc1);
        const f32x8 acc = acc0 + acc1;
        #pragma unroll
        for (int j = 0; j < 8; ++j) {
            const int tok = mb * 16 + j + hi8;
            const int c   = hh * HD + db * 16 + nn;
            ctx[tok * CS + c] = (bf16)acc[j];
        }
    }
    __syncthreads();

    // ---- out = ctx @ Wproj^T + b : 4 x 12 tiles, K=192 -------------------
    float* og = out + (size_t)w * (Nt * C);
    for (int t = wv; t < 4 * 12; t += 8) {
        const int mb = t / 12, nb = t % 12;
        f32x8 acc0 = {}, acc1 = {};
        #pragma unroll
        for (int k0 = 0; k0 < C; k0 += 64) {
            acc0 = wmma_bf16(load_frag_a(ctx + mb * 16 * CS + k0, CS, lane),
                             load_frag_b(wproj + (size_t)(nb * 16) * C + k0, C, lane), acc0);
            acc1 = wmma_bf16(load_frag_a(ctx + mb * 16 * CS + k0 + 32, CS, lane),
                             load_frag_b(wproj + (size_t)(nb * 16) * C + k0 + 32, C, lane), acc1);
        }
        const f32x8 acc = acc0 + acc1;
        const int c = nb * 16 + nn;
        const float bias = bproj[c];
        #pragma unroll
        for (int j = 0; j < 8; ++j) {
            const int tok = mb * 16 + j + hi8;
            og[tok * C + c] = acc[j] + bias;
        }
    }
}

// ---------------------------------------------------------------------------
extern "C" void kernel_launch(void* const* d_in, const int* in_sizes, int n_in,
                              void* d_out, int out_size, void* d_ws, size_t ws_size,
                              hipStream_t stream) {
    (void)in_sizes; (void)n_in; (void)out_size; (void)ws_size;
    const float* x      = (const float*)d_in[0];   // [4096,64,192]
    const float* mask   = (const float*)d_in[1];   // [4096,64,64]
    const float* qkv_w  = (const float*)d_in[2];   // [576,192]
    const float* qkv_b  = (const float*)d_in[3];   // [576]
    const float* proj_w = (const float*)d_in[4];   // [192,192]
    const float* proj_b = (const float*)d_in[5];   // [192]
    float* out = (float*)d_out;                    // [4096,64,192]

    bf16* wq_bf = (bf16*)d_ws;                     // 576*192 bf16
    bf16* wp_bf = wq_bf + C3 * C;                  // 192*192 bf16

    wa_convert_weights<<<(C3 * C + 255) / 256, 256, 0, stream>>>(qkv_w, proj_w, wq_bf, wp_bf);
    wa_window_attention<<<4096, 256, LDS_BYTES, stream>>>(
        x, mask, wq_bf, qkv_b, wp_bf, proj_b, out);
}
